// Sc_encoder_73108933312980
// MI455X (gfx1250) — matrix operand order repaired and checked by reading, not compiled, not run on hardware
//
#include <hip/hip_runtime.h>

// ---------------------------------------------------------------------------
// Types for CDNA5 WMMA (gfx1250, wave32)
// ---------------------------------------------------------------------------
typedef __bf16 bf16_t;
typedef bf16_t v16bf __attribute__((ext_vector_type(16)));
typedef bf16_t v8bf  __attribute__((ext_vector_type(8)));
typedef float  v8f   __attribute__((ext_vector_type(8)));

#define HID 128

// gfx1250 async global->LDS path (ASYNCcnt). Probe-confirmed: the builtin's
// first param is 'int __vector(4) __device__*' (AS1), second is AS3.
#if defined(__has_builtin)
#  if __has_builtin(__builtin_amdgcn_global_load_async_to_lds_b128) && \
      __has_builtin(__builtin_amdgcn_s_wait_asynccnt)
#    define HAVE_ASYNC_LDS 1
#  endif
#endif

#if defined(HAVE_ASYNC_LDS)
typedef int i32x4 __attribute__((vector_size(16)));
typedef __attribute__((address_space(1))) i32x4* gas_i32x4_p;
typedef __attribute__((address_space(3))) i32x4* lds_i32x4_p;
#endif

static __device__ __forceinline__ unsigned short f2bf(float f) {
    unsigned u = __float_as_uint(f);
    unsigned r = u + 0x7FFFu + ((u >> 16) & 1u);       // round-to-nearest-even
    if ((u & 0x7F800000u) == 0x7F800000u) r = u;       // inf/nan: truncate
    return (unsigned short)(r >> 16);
}

// monotonic float<->uint keys so integer atomicMax implements float max
static __device__ __forceinline__ unsigned fkey(float f) {
    unsigned u = __float_as_uint(f);
    return (u & 0x80000000u) ? ~u : (u | 0x80000000u);
}
static __device__ __forceinline__ float funkey(unsigned k) {
    return __uint_as_float((k & 0x80000000u) ? (k & 0x7FFFFFFFu) : ~k);
}

// ---------------------------------------------------------------------------
// Elementwise helpers
// ---------------------------------------------------------------------------
__global__ void k_f32_to_bf16(const float* __restrict__ in,
                              unsigned short* __restrict__ out, int n) {
    int i = blockIdx.x * blockDim.x + threadIdx.x;
    if (i < n) out[i] = f2bf(in[i]);
}

__global__ void k_fill_u32(unsigned* __restrict__ p, unsigned v, int n) {
    int i = blockIdx.x * blockDim.x + threadIdx.x;
    if (i < n) p[i] = v;
}

__global__ void k_relu(const float* __restrict__ in, float* __restrict__ out, int n) {
    int i = blockIdx.x * blockDim.x + threadIdx.x;
    if (i < n) out[i] = fmaxf(in[i], 0.0f);
}

// agg[i] = sum of biases of relations targeting this node type (handles
// isolated dst nodes too; GAT adds bias unconditionally).
__global__ void k_init_agg(float* __restrict__ agg, const float* __restrict__ bias,
                           int base, int r0, int r1, int n) {
    int i = blockIdx.x * blockDim.x + threadIdx.x;
    if (i >= n) return;
    int k = i & (HID - 1);
    float b = bias[base + r0 * HID + k];
    if (r1 >= 0) b += bias[base + r1 * HID + k];
    agg[i] = b;
}

// c[b][k] = sum_n W[b][n][k] * att[b][n]   (collapses hd @ a_d into x @ c)
__global__ void k_compute_c(const float* __restrict__ W, const float* __restrict__ att,
                            float* __restrict__ c) {
    int b = blockIdx.x;     // layer*8 + r   (16 blocks)
    int k = threadIdx.x;    // 128 threads
    const float* Wb = W + b * (HID * HID);
    const float* ab = att + b * HID;
    float s = 0.0f;
    #pragma unroll 8
    for (int n = 0; n < HID; ++n) s += Wb[n * HID + k] * ab[n];
    c[b * HID + k] = s;
}

// al[i] = dot(x[i,:], c)  — one wave per node, fp32, shfl reduction
__global__ __launch_bounds__(256) void k_node_alpha(const float* __restrict__ x,
                                                    const float* __restrict__ c,
                                                    float* __restrict__ al, int n) {
    int wid  = (blockIdx.x * blockDim.x + threadIdx.x) >> 5;
    int lane = threadIdx.x & 31;
    if (wid >= n) return;
    float4 xv = *(const float4*)(x + (size_t)wid * HID + lane * 4);
    float4 cv = *(const float4*)(c + lane * 4);
    float p = xv.x * cv.x + xv.y * cv.y + xv.z * cv.z + xv.w * cv.w;
    #pragma unroll
    for (int off = 16; off; off >>= 1) p += __shfl_xor(p, off, 32);
    if (lane == 0) al[wid] = p;
}

// ---------------------------------------------------------------------------
// hs = A[M,128](bf16) @ W[128,128](bf16, row n / col k)^T  -> C[M,128] f32
// Block tile: 64 rows. 8 waves each own a 16-col stripe; each wave computes
// 4 independent 16x16 M-subtiles (4 independent WMMA accumulation chains,
// hides XDL latency; B fragment reused 4x per k-step). K looped by 32.
// A and B fragments both use contiguous k: element s -> k0 + 16*hl + s (wave32).
// ---------------------------------------------------------------------------
__global__ __launch_bounds__(256) void k_gemm_proj_bf16(
        const unsigned short* __restrict__ A,
        const unsigned short* __restrict__ W,
        float* __restrict__ C, int M) {
    __shared__ __align__(16) unsigned short As[64 * HID];   // 16 KB A tile
    int m0 = blockIdx.x * 64;
    // stage 64x128 bf16 tile: 1024 x 16B, 256 threads x 4. Rows are clamped so
    // a partial last tile (M is always a multiple of 16) stays in bounds.
    #pragma unroll
    for (int i = threadIdx.x; i < 1024; i += 256) {
        int gr = m0 + (i >> 4);
        if (gr >= M) gr = M - 1;
        const uint4* g = (const uint4*)(A + (size_t)gr * HID) + (i & 15);
#if defined(HAVE_ASYNC_LDS)
        __builtin_amdgcn_global_load_async_to_lds_b128(
            (gas_i32x4_p)g, (lds_i32x4_p)&As[i * 8], 0, 0);
#else
        ((uint4*)As)[i] = *g;
#endif
    }
#if defined(HAVE_ASYNC_LDS)
    __builtin_amdgcn_s_wait_asynccnt(0);
#endif
    __syncthreads();

    int lane = threadIdx.x & 31;
    int wv   = threadIdx.x >> 5;
    int n0   = wv * 16;
    int hl   = lane >> 4;        // half-wave select
    int l16  = lane & 15;

    v8f acc[4];
    #pragma unroll
    for (int s = 0; s < 4; ++s) acc[s] = (v8f){0.f,0.f,0.f,0.f,0.f,0.f,0.f,0.f};

    const bf16_t* abase = (const bf16_t*)As + l16 * HID;
    const bf16_t* brow  = (const bf16_t*)W + (n0 + l16) * HID + hl * 16; // col n=n0+l16

    #pragma unroll
    for (int k0 = 0; k0 < HID; k0 += 32) {
        // B 32x16 bf16: lane holds 16 consecutive K of column n (contiguous in W row)
        v16bf b = *(const v16bf*)(brow + k0);
        #pragma unroll
        for (int s = 0; s < 4; ++s) {
            // A 16x32 bf16: lanes<16 hold K = k0+{0..15}; lanes>=16: k0+{16..31}
            const bf16_t* arow = abase + s * 16 * HID;
            v8bf alo = *(const v8bf*)(arow + k0 + hl * 16);
            v8bf ahi = *(const v8bf*)(arow + k0 + hl * 16 + 8);
            v16bf a = __builtin_shufflevector(alo, ahi,
                        0,1,2,3,4,5,6,7,8,9,10,11,12,13,14,15);
            acc[s] = __builtin_amdgcn_wmma_f32_16x16x32_bf16(
                         false, a, false, b, (short)0, acc[s], false, false);
        }
    }
    // C 16x16 f32: VGPR j -> row (subtile base) + j + hl*8, col n0 + l16.
    // Subtile bounds are multiples of 16, M is a multiple of 16 -> guard is
    // wave-uniform (EXEC stayed all-ones through the WMMAs above).
    #pragma unroll
    for (int s = 0; s < 4; ++s) {
        int mrow = m0 + s * 16;
        if (mrow >= M) break;
        float* crow = C + (size_t)(mrow + hl * 8) * HID + n0 + l16;
        #pragma unroll
        for (int j = 0; j < 8; ++j) crow[(size_t)j * HID] = acc[s][j];
    }
}

// ---------------------------------------------------------------------------
// Edge passes (softmax over dst segments)
// ---------------------------------------------------------------------------
__global__ void k_edge_max(const int* __restrict__ src, const int* __restrict__ dst,
                           const float* __restrict__ als, const float* __restrict__ ald,
                           unsigned* __restrict__ mkey, int E) {
    int i = blockIdx.x * blockDim.x + threadIdx.x;
    if (i >= E) return;
    float e = als[src[i]] + ald[dst[i]];
    e = e >= 0.f ? e : 0.2f * e;
    atomicMax(&mkey[dst[i]], fkey(e));
}

__global__ void k_edge_expsum(const int* __restrict__ src, const int* __restrict__ dst,
                              const float* __restrict__ als, const float* __restrict__ ald,
                              const unsigned* __restrict__ mkey, float* __restrict__ denom,
                              int E) {
    int i = blockIdx.x * blockDim.x + threadIdx.x;
    if (i >= E) return;
    int d = dst[i];
    float e = als[src[i]] + ald[d];
    e = e >= 0.f ? e : 0.2f * e;
    unsafeAtomicAdd(&denom[d], expf(e - funkey(mkey[d])));
}

// one wave per edge: scatter 128-f32 row with global_atomic_add_f32
__global__ __launch_bounds__(256) void k_edge_agg(
        const int* __restrict__ src, const int* __restrict__ dst,
        const float* __restrict__ als, const float* __restrict__ ald,
        const unsigned* __restrict__ mkey, const float* __restrict__ denom,
        const float* __restrict__ hs, float* __restrict__ agg, int E) {
    int wid  = (blockIdx.x * blockDim.x + threadIdx.x) >> 5;
    int lane = threadIdx.x & 31;
    if (wid >= E) return;
    int s = src[wid], d = dst[wid];
    float e = als[s] + ald[d];
    e = e >= 0.f ? e : 0.2f * e;
    float coef = expf(e - funkey(mkey[d])) / (denom[d] + 1e-16f);
    float4 h = *(const float4*)(hs + (size_t)s * HID + lane * 4);
    float* a = agg + (size_t)d * HID + lane * 4;
    unsafeAtomicAdd(a + 0, h.x * coef);
    unsafeAtomicAdd(a + 1, h.y * coef);
    unsafeAtomicAdd(a + 2, h.z * coef);
    unsafeAtomicAdd(a + 3, h.w * coef);
}

// ---------------------------------------------------------------------------
// Orchestration
// ---------------------------------------------------------------------------
static inline int cdiv(long a, long b) { return (int)((a + b - 1) / b); }

extern "C" void kernel_launch(void* const* d_in, const int* in_sizes, int n_in,
                              void* d_out, int out_size, void* d_ws, size_t ws_size,
                              hipStream_t stream) {
    (void)n_in; (void)out_size; (void)ws_size;
    // node types: 0=datacenter 1=host 2=vm 3=task 4=instance
    int   ncnt[5];
    long  off[6]; off[0] = 0;
    for (int t = 0; t < 5; ++t) { ncnt[t] = in_sizes[t] / HID; off[t + 1] = off[t] + ncnt[t]; }
    long TOT = off[5];
    int maxN = 0; for (int t = 0; t < 5; ++t) maxN = ncnt[t] > maxN ? ncnt[t] : maxN;
    int Er[8]; for (int r = 0; r < 8; ++r) Er[r] = in_sizes[5 + r] / 2;

    const float* W_src   = (const float*)d_in[13];
    const float* W_dst   = (const float*)d_in[14];
    const float* att_src = (const float*)d_in[15];
    const float* att_dst = (const float*)d_in[16];
    const float* bias    = (const float*)d_in[17];

    // scratch carve
    char* w = (char*)d_ws;
    auto carve = [&](size_t bytes) -> char* {
        char* p = w; w += (bytes + 255) & ~(size_t)255; return p;
    };
    float*          xbuf  = (float*)carve((size_t)TOT * HID * 4);
    float*          agg   = (float*)carve((size_t)TOT * HID * 4);
    unsigned short* xb    = (unsigned short*)carve((size_t)TOT * HID * 2);
    float*          hs    = (float*)carve((size_t)maxN * HID * 4);
    float*          als   = (float*)carve((size_t)maxN * 4);
    float*          ald   = (float*)carve((size_t)maxN * 4);
    unsigned*       mkey  = (unsigned*)carve((size_t)maxN * 4);
    float*          denom = (float*)carve((size_t)maxN * 4);
    unsigned short* Wb    = (unsigned short*)carve((size_t)16 * HID * HID * 2);
    float*          cs    = (float*)carve((size_t)16 * HID * 4);
    float*          cd    = (float*)carve((size_t)16 * HID * 4);

    // one-time precompute
    { int n = 16 * HID * HID;
      k_f32_to_bf16<<<cdiv(n, 256), 256, 0, stream>>>(W_src, Wb, n); }
    k_compute_c<<<16, HID, 0, stream>>>(W_src, att_src, cs);
    k_compute_c<<<16, HID, 0, stream>>>(W_dst, att_dst, cd);

    static const int rs[8] = {0, 0, 1, 2, 3, 3, 4, 2};   // src type per relation
    static const int rd[8] = {1, 2, 0, 0, 3, 4, 2, 1};   // dst type per relation
    static const int br0[5] = {2, 0, 1, 4, 5};           // relations targeting each type
    static const int br1[5] = {3, 7, 6, -1, -1};

    const float* xin[5];
    for (int t = 0; t < 5; ++t) xin[t] = (const float*)d_in[t];

    for (int layer = 0; layer < 2; ++layer) {
        // bf16 copies of node features for WMMA (only types used as sources)
        bool src_used[5] = {true, layer == 0, true, true, true};
        for (int t = 0; t < 5; ++t) {
            if (!src_used[t]) continue;
            int n = ncnt[t] * HID;
            k_f32_to_bf16<<<cdiv(n, 256), 256, 0, stream>>>(xin[t], xb + off[t] * HID, n);
        }
        // layer 2 only needs dst in {host, vm, instance}; dc/task outputs are dead
        bool active[5] = {layer == 0, true, true, layer == 0, true};
        for (int t = 0; t < 5; ++t) {
            if (!active[t]) continue;
            int n = ncnt[t] * HID;
            k_init_agg<<<cdiv(n, 256), 256, 0, stream>>>(
                agg + off[t] * HID, bias, layer * 8 * HID, br0[t], br1[t], n);
        }
        static const int rl0[8] = {0, 1, 2, 3, 4, 5, 6, 7};
        static const int rl1[5] = {0, 1, 5, 6, 7};
        const int* rlist = layer ? rl1 : rl0;
        int nr = layer ? 5 : 8;

        for (int ri = 0; ri < nr; ++ri) {
            int r = rlist[ri], s = rs[r], d = rd[r];
            int Ns = ncnt[s], Nd = ncnt[d], E = Er[r];
            const int* ei   = (const int*)d_in[5 + r];
            const int* esrc = ei;
            const int* edst = ei + E;
            int cb = (layer * 8 + r) * HID;

            k_gemm_proj_bf16<<<cdiv(Ns, 64), 256, 0, stream>>>(
                xb + off[s] * HID, Wb + (size_t)(layer * 8 + r) * HID * HID, hs, Ns);
            k_node_alpha<<<cdiv((long)Ns * 32, 256), 256, 0, stream>>>(xin[s], cs + cb, als, Ns);
            k_node_alpha<<<cdiv((long)Nd * 32, 256), 256, 0, stream>>>(xin[d], cd + cb, ald, Nd);
            k_fill_u32<<<cdiv(Nd, 256), 256, 0, stream>>>(mkey, 0u, Nd);
            k_fill_u32<<<cdiv(Nd, 256), 256, 0, stream>>>((unsigned*)denom, 0u, Nd);
            k_edge_max<<<cdiv(E, 256), 256, 0, stream>>>(esrc, edst, als, ald, mkey, E);
            k_edge_expsum<<<cdiv(E, 256), 256, 0, stream>>>(esrc, edst, als, ald, mkey, denom, E);
            k_edge_agg<<<cdiv((long)E * 32, 256), 256, 0, stream>>>(
                esrc, edst, als, ald, mkey, denom, hs, agg + off[d] * HID, E);
        }

        if (layer == 0) {
            for (int t = 0; t < 5; ++t) {
                int n = ncnt[t] * HID;
                k_relu<<<cdiv(n, 256), 256, 0, stream>>>(agg + off[t] * HID, xbuf + off[t] * HID, n);
                xin[t] = xbuf + off[t] * HID;
            }
        } else {
            // inter-attention beta == softmax(single logit) == 1.0 exactly:
            // outputs are relu(agg) for host, vm, instance, concatenated.
            float* out = (float*)d_out;
            long oh = 0, ov = (long)ncnt[1] * HID, oi = (long)(ncnt[1] + ncnt[2]) * HID;
            k_relu<<<cdiv((long)ncnt[1] * HID, 256), 256, 0, stream>>>(agg + off[1] * HID, out + oh, ncnt[1] * HID);
            k_relu<<<cdiv((long)ncnt[2] * HID, 256), 256, 0, stream>>>(agg + off[2] * HID, out + ov, ncnt[2] * HID);
            k_relu<<<cdiv((long)ncnt[4] * HID, 256), 256, 0, stream>>>(agg + off[4] * HID, out + oi, ncnt[4] * HID);
        }
    }
}
